// EMPSNLayer_43035572306254
// MI455X (gfx1250) — compile-verified, matrix operand order checked by reading, not run on hardware
//
#include <hip/hip_runtime.h>
#include <hip/hip_bf16.h>

#define HD 128

typedef __attribute__((ext_vector_type(16))) _Float16 v16h;
typedef __attribute__((ext_vector_type(8)))  float    v8f;
typedef __attribute__((ext_vector_type(4)))  _Float16 v4h;

union F16x16 { v16h v; float4 q[2]; };

static constexpr int kN0 = 20000, kN1 = 60000, kN2 = 30000, kE = 200000;

// ---------------------------------------------------------------------------
// f32 -> f16 elementwise convert (vectorized x4)
// ---------------------------------------------------------------------------
__global__ __launch_bounds__(256) void f32_to_f16_kernel(
    const float* __restrict__ src, _Float16* __restrict__ dst, long n4)
{
    long i = (long)blockIdx.x * blockDim.x + threadIdx.x;
    if (i >= n4) return;
    float4 v = *(const float4*)(src + i * 4);
    v4h h = {(_Float16)v.x, (_Float16)v.y, (_Float16)v.z, (_Float16)v.w};
    *(v4h*)(dst + i * 4) = h;
}

// ---------------------------------------------------------------------------
// Pack f32 weights [din][128] (row-major, K-major) into WMMA B-fragment order:
// out[((kc*8 + nt)*32 + lane)*16 + j], zero-padded for k >= din.
// Lane L: column n = nt*16 + (L&15), h = L>>4;
// halves j<8 -> k = kc*32 + 8h + j ; j>=8 -> k = kc*32 + 16 + 8h + (j-8)
// ---------------------------------------------------------------------------
__global__ __launch_bounds__(256) void pack_w_kernel(
    const float* __restrict__ W, _Float16* __restrict__ out, int din, int KC)
{
    int idx = blockIdx.x * blockDim.x + threadIdx.x;
    int total = KC * 8 * 32 * 16;
    if (idx >= total) return;
    int j  = idx & 15;
    int L  = (idx >> 4) & 31;
    int nt = (idx >> 9) & 7;
    int kc = idx >> 12;
    int n  = nt * 16 + (L & 15);
    int h  = L >> 4;
    int kl = (j < 8) ? (8 * h + j) : (16 + 8 * h + (j - 8));
    int k  = kc * 32 + kl;
    out[idx] = (_Float16)((k < din) ? W[(size_t)k * HD + n] : 0.f);
}

// ---------------------------------------------------------------------------
// Shared WMMA tile compute: 8 waves, 64 rows x 128 cols per block.
// Wave w: row-tile rt = w&3 (16 rows), col-half c = w>>2 (64 cols, 4 n-tiles).
// A frags from LDS (stride padded), B frags from pre-packed global (L2 hot).
// Stores z (f32) and accumulates per-feature sum / sum-of-squares for BN.
// ---------------------------------------------------------------------------
template<int KC>
__device__ __forceinline__ void tile_mma(
    const _Float16* __restrict__ sA, float* __restrict__ sStats,
    const _Float16* __restrict__ Wp, float* __restrict__ z,
    long rowBase, int nRows)
{
    constexpr int STRIDE = KC * 32 + 8;
    const int t    = threadIdx.x;
    const int wave = t >> 5, lane = t & 31;
    const int rt   = wave & 3, c = wave >> 2;
    const int m    = lane & 15, hh = lane >> 4;

    const v8f zero8 = {0.f, 0.f, 0.f, 0.f, 0.f, 0.f, 0.f, 0.f};
    v8f acc[4];
#pragma unroll
    for (int n = 0; n < 4; ++n) acc[n] = zero8;

    const _Float16* rowp = sA + (size_t)(rt * 16 + m) * STRIDE;
#pragma unroll
    for (int kc = 0; kc < KC; ++kc) {
        F16x16 a;
        a.q[0] = *(const float4*)(rowp + kc * 32 + 8 * hh);
        a.q[1] = *(const float4*)(rowp + kc * 32 + 16 + 8 * hh);
        const _Float16* wp = Wp + ((size_t)((kc * 8 + c * 4) * 32 + lane)) * 16;
#pragma unroll
        for (int n = 0; n < 4; ++n) {
            F16x16 b;
            const float4* bq = (const float4*)(wp + n * (32 * 16));
            b.q[0] = bq[0];
            b.q[1] = bq[1];
            acc[n] = __builtin_amdgcn_wmma_f32_16x16x32_f16(
                false, a.v, false, b.v, (short)0, acc[n], false, false);
        }
    }

#pragma unroll
    for (int n = 0; n < 4; ++n) {
        const int col = c * 64 + n * 16 + m;
        float s1 = 0.f, s2 = 0.f;
#pragma unroll
        for (int r = 0; r < 8; ++r) {
            float v = acc[n][r];
            s1 += v;
            s2 += v * v;
            long rg = rowBase + rt * 16 + hh * 8 + r;
            if (rg < nRows) z[(size_t)rg * HD + col] = v;
        }
        s1 += __shfl_xor(s1, 16);
        s2 += __shfl_xor(s2, 16);
        if (hh == 0) {
            atomicAdd(&sStats[col], s1);
            atomicAdd(&sStats[128 + col], s2);
        }
    }
}

// ---------------------------------------------------------------------------
// Message GEMM: rows = gathered [x_s[i_s] | x_r[i_r] | inv(pad)] in f16.
// ---------------------------------------------------------------------------
template<int KC>
__global__ __launch_bounds__(256) void msg_gemm(
    const _Float16* __restrict__ xs, const _Float16* __restrict__ xr,
    const int* __restrict__ i_s, const int* __restrict__ i_r,
    const float* __restrict__ inv, int ninv,
    const _Float16* __restrict__ Wp,
    float* __restrict__ z, float* __restrict__ gstats, int nE)
{
    constexpr int STRIDE = KC * 32 + 8;
    __shared__ _Float16 sA[64 * STRIDE];
    __shared__ float sStats[256];

    const int t = threadIdx.x;
    sStats[t] = 0.f;

    const int row = t >> 2, part = t & 3;
    const long e = (long)blockIdx.x * 64 + row;
    _Float16* dst = sA + (size_t)row * STRIDE;
    const float4 z4 = make_float4(0.f, 0.f, 0.f, 0.f);

    if (e < nE) {
        if (part < 2) {
            const _Float16* src = xs + (size_t)i_s[e] * HD + part * 64;
#pragma unroll
            for (int i = 0; i < 8; ++i)
                *(float4*)(dst + part * 64 + i * 8) = *(const float4*)(src + i * 8);
        } else {
            const _Float16* src = xr + (size_t)i_r[e] * HD + (part - 2) * 64;
#pragma unroll
            for (int i = 0; i < 8; ++i)
                *(float4*)(dst + 128 + (part - 2) * 64 + i * 8) = *(const float4*)(src + i * 8);
            if (part == 3) {
#pragma unroll
                for (int i = 0; i < 32; ++i)
                    dst[256 + i] = (_Float16)((i < ninv) ? inv[(size_t)e * ninv + i] : 0.f);
            }
        }
    } else {
#pragma unroll
        for (int i = 0; i < 8; ++i)
            *(float4*)(dst + part * 64 + i * 8) = z4;
        if (part == 3) {
#pragma unroll
            for (int i = 0; i < 32; ++i) dst[256 + i] = (_Float16)0.f;
        }
    }
    __syncthreads();

    tile_mma<KC>(sA, sStats, Wp, z, (long)blockIdx.x * 64, nE);

    __syncthreads();
    atomicAdd(&gstats[t], sStats[t]);
}

// ---------------------------------------------------------------------------
// Message finalize: BN (from global stats) -> SiLU -> edge gate -> scatter-add.
// One wave per edge; lane covers 4 strided features.
// ---------------------------------------------------------------------------
__global__ __launch_bounds__(256) void msg_finalize(
    const float* __restrict__ z, const float* __restrict__ gstats,
    const float* __restrict__ g1, const float* __restrict__ bt1,
    const float* __restrict__ We, const float* __restrict__ be,
    const int* __restrict__ i_r, float* __restrict__ mes, int nE)
{
    const int wave = threadIdx.x >> 5, lane = threadIdx.x & 31;
    const long e = (long)blockIdx.x * 8 + wave;
    if (e >= nE) return;

    const float invN = 1.f / (float)nE;
    const int r = i_r[e];
    float acts[4];
    float partial = 0.f;
#pragma unroll
    for (int i = 0; i < 4; ++i) {
        const int f = lane + 32 * i;
        float mu  = gstats[f] * invN;
        float var = gstats[128 + f] * invN - mu * mu;
        float sf  = g1[f] * rsqrtf(var + 1e-5f);
        float tf  = bt1[f] - mu * sf;
        float mm  = z[(size_t)e * HD + f] * sf + tf;
        float act = mm / (1.f + __expf(-mm));   // SiLU
        acts[i] = act;
        partial += act * We[f];
    }
#pragma unroll
    for (int off = 16; off > 0; off >>= 1) partial += __shfl_xor(partial, off);
    const float wg = 1.f / (1.f + __expf(-(partial + be[0])));
#pragma unroll
    for (int i = 0; i < 4; ++i) {
        const int f = lane + 32 * i;
        atomicAdd(&mes[(size_t)r * HD + f], acts[i] * wg);
    }
}

// ---------------------------------------------------------------------------
// Update GEMM: rows = [x_d(f16) | mes_a1(f32->f16) | (mes_a2)] ; K = NS*128+128.
// ---------------------------------------------------------------------------
template<int KC, int NS>
__global__ __launch_bounds__(256) void upd_gemm(
    const _Float16* __restrict__ xh, const float* __restrict__ m1,
    const float* __restrict__ m2, const _Float16* __restrict__ Wp,
    float* __restrict__ z, float* __restrict__ gstats, int nRows)
{
    constexpr int STRIDE = KC * 32 + 8;
    __shared__ _Float16 sA[64 * STRIDE];
    __shared__ float sStats[256];

    const int t = threadIdx.x;
    sStats[t] = 0.f;

    const int row = t >> 2, part = t & 3;
    const long rg = (long)blockIdx.x * 64 + row;
    const bool valid = rg < nRows;
    _Float16* dst = sA + (size_t)row * STRIDE;
    const float4 z4 = make_float4(0.f, 0.f, 0.f, 0.f);

    {   // x features, already f16: 32 halves per part
        const _Float16* src = xh + (size_t)rg * HD + part * 32;
#pragma unroll
        for (int i = 0; i < 4; ++i)
            *(float4*)(dst + part * 32 + i * 8) = valid ? *(const float4*)(src + i * 8) : z4;
    }
    {   // mes1: f32 -> f16, 32 elems per part
        const float* src = m1 + (size_t)rg * HD + part * 32;
#pragma unroll
        for (int i = 0; i < 8; ++i) {
            float4 v = valid ? *(const float4*)(src + i * 4) : z4;
            v4h h = {(_Float16)v.x, (_Float16)v.y, (_Float16)v.z, (_Float16)v.w};
            *(v4h*)(dst + 128 + part * 32 + i * 4) = h;
        }
    }
    if (NS == 2) {
        const float* src = m2 + (size_t)rg * HD + part * 32;
#pragma unroll
        for (int i = 0; i < 8; ++i) {
            float4 v = valid ? *(const float4*)(src + i * 4) : z4;
            v4h h = {(_Float16)v.x, (_Float16)v.y, (_Float16)v.z, (_Float16)v.w};
            *(v4h*)(dst + 256 + part * 32 + i * 4) = h;
        }
    }
    __syncthreads();

    tile_mma<KC>(sA, sStats, Wp, z, (long)blockIdx.x * 64, nRows);

    __syncthreads();
    atomicAdd(&gstats[t], sStats[t]);
}

// ---------------------------------------------------------------------------
// Update finalize: out = x + BN(z2)  (bias cancels inside BN)
// ---------------------------------------------------------------------------
__global__ __launch_bounds__(256) void upd_finalize(
    const float* __restrict__ x, const float* __restrict__ z,
    const float* __restrict__ gstats, const float* __restrict__ g,
    const float* __restrict__ bt, float* __restrict__ out, int nRows)
{
    const int wave = threadIdx.x >> 5, lane = threadIdx.x & 31;
    const long i = (long)blockIdx.x * 8 + wave;
    if (i >= nRows) return;
    const float invN = 1.f / (float)nRows;
#pragma unroll
    for (int k = 0; k < 4; ++k) {
        const int f = lane + 32 * k;
        float mu  = gstats[f] * invN;
        float var = gstats[128 + f] * invN - mu * mu;
        float sf  = g[f] * rsqrtf(var + 1e-5f);
        float tf  = bt[f] - mu * sf;
        out[(size_t)i * HD + f] = x[(size_t)i * HD + f] + z[(size_t)i * HD + f] * sf + tf;
    }
}

// ---------------------------------------------------------------------------
// Host launcher
// ---------------------------------------------------------------------------
extern "C" void kernel_launch(void* const* d_in, const int* in_sizes, int n_in,
                              void* d_out, int out_size, void* d_ws, size_t ws_size,
                              hipStream_t stream)
{
    (void)in_sizes; (void)n_in; (void)out_size; (void)ws_size;

    // --- inputs (setup_inputs dict order) ---
    const float* x[3] = {(const float*)d_in[0], (const float*)d_in[1], (const float*)d_in[2]};
    const int Ndim[3] = {kN0, kN1, kN2};
    const int ninvA[5] = {3, 3, 6, 6, 6};
    const int sdim[5]  = {0, 0, 1, 1, 2};
    const int* isA[5]; const int* irA[5]; const float* invA[5];
    for (int a = 0; a < 5; ++a) {
        isA[a]  = (const int*)d_in[3 + 3 * a];
        irA[a]  = (const int*)d_in[4 + 3 * a];
        invA[a] = (const float*)d_in[5 + 3 * a];
    }
    // params flattened alphabetically: mp{a}{W1,We,b1,be,bt1,g1}, up{d}{W,b,bt,g}
    const float *W1[5], *We[5], *be[5], *bt1[5], *g1[5];
    for (int a = 0; a < 5; ++a) {
        const int base = 18 + 6 * a;
        W1[a]  = (const float*)d_in[base + 0];
        We[a]  = (const float*)d_in[base + 1];
        // b1 (base+2) cancels inside BN
        be[a]  = (const float*)d_in[base + 3];
        bt1[a] = (const float*)d_in[base + 4];
        g1[a]  = (const float*)d_in[base + 5];
    }
    const float *Wu[3], *btu[3], *gu[3];
    for (int d = 0; d < 3; ++d) {
        const int base = 48 + 4 * d;
        Wu[d]  = (const float*)d_in[base + 0];
        // b (base+1) cancels inside BN
        btu[d] = (const float*)d_in[base + 2];
        gu[d]  = (const float*)d_in[base + 3];
    }

    // --- workspace carve ---
    char* p = (char*)d_ws;
    auto carve = [&](size_t bytes) -> char* {
        char* r = p;
        p += (bytes + 255) & ~(size_t)255;
        return r;
    };
    _Float16* xh[3];
    for (int d = 0; d < 3; ++d) xh[d] = (_Float16*)carve((size_t)Ndim[d] * HD * 2);
    _Float16* WpMsg[5];
    for (int a = 0; a < 5; ++a) WpMsg[a] = (_Float16*)carve((size_t)9 * 4096 * 2);
    const int KCu[3] = {8, 12, 12};
    _Float16* WpUpd[3];
    for (int d = 0; d < 3; ++d) WpUpd[d] = (_Float16*)carve((size_t)KCu[d] * 4096 * 2);
    float* zbuf = (float*)carve((size_t)kE * HD * 4);
    const int mesRows[5] = {kN0, kN1, kN1, kN2, kN2};
    float* mes[5];
    for (int a = 0; a < 5; ++a) mes[a] = (float*)carve((size_t)mesRows[a] * HD * 4);
    float* stats = (float*)carve(8 * 256 * 4);

    // --- zero accumulators ---
    hipMemsetAsync(stats, 0, 8 * 256 * 4, stream);
    for (int a = 0; a < 5; ++a)
        hipMemsetAsync(mes[a], 0, (size_t)mesRows[a] * HD * 4, stream);

    // --- convert x to f16 ---
    for (int d = 0; d < 3; ++d) {
        long n4 = (long)Ndim[d] * HD / 4;
        f32_to_f16_kernel<<<(int)((n4 + 255) / 256), 256, 0, stream>>>(x[d], xh[d], n4);
    }
    // --- pack weights into B-fragment order ---
    for (int a = 0; a < 5; ++a)
        pack_w_kernel<<<144, 256, 0, stream>>>(W1[a], WpMsg[a], 256 + ninvA[a], 9);
    for (int d = 0; d < 3; ++d)
        pack_w_kernel<<<KCu[d] * 16, 256, 0, stream>>>(Wu[d], WpUpd[d], KCu[d] == 8 ? 256 : 384, KCu[d]);

    // --- message passing per adjacency ---
    const int msgBlocks = (kE + 63) / 64;
    const int finBlocks = (kE + 7) / 8;
    for (int a = 0; a < 5; ++a) {
        float* gst = stats + a * 256;
        const int rd = (a == 0) ? 0 : (a <= 2 ? 1 : 2); // receiver dims: 0,1,1,2,2
        msg_gemm<9><<<msgBlocks, 256, 0, stream>>>(
            xh[sdim[a]], xh[rd], isA[a], irA[a], invA[a], ninvA[a],
            WpMsg[a], zbuf, gst, kE);
        msg_finalize<<<finBlocks, 256, 0, stream>>>(
            zbuf, gst, g1[a], bt1[a], We[a], be[a], irA[a], mes[a], kE);
    }

    // --- updates per node type ---
    float* out0 = (float*)d_out;
    float* out1 = out0 + (size_t)kN0 * HD;
    float* out2 = out1 + (size_t)kN1 * HD;

    upd_gemm<8, 1><<<(kN0 + 63) / 64, 256, 0, stream>>>(
        xh[0], mes[0], nullptr, WpUpd[0], zbuf, stats + 5 * 256, kN0);
    upd_finalize<<<(kN0 + 7) / 8, 256, 0, stream>>>(
        x[0], zbuf, stats + 5 * 256, gu[0], btu[0], out0, kN0);

    upd_gemm<12, 2><<<(kN1 + 63) / 64, 256, 0, stream>>>(
        xh[1], mes[1], mes[2], WpUpd[1], zbuf, stats + 6 * 256, kN1);
    upd_finalize<<<(kN1 + 7) / 8, 256, 0, stream>>>(
        x[1], zbuf, stats + 6 * 256, gu[1], btu[1], out1, kN1);

    upd_gemm<12, 2><<<(kN2 + 63) / 64, 256, 0, stream>>>(
        xh[2], mes[3], mes[4], WpUpd[2], zbuf, stats + 7 * 256, kN2);
    upd_finalize<<<(kN2 + 7) / 8, 256, 0, stream>>>(
        x[2], zbuf, stats + 7 * 256, gu[2], btu[2], out2, kN2);
}
